// OptimizedSEA_Attention_36515811951138
// MI455X (gfx1250) — matrix operand order repaired and checked by reading, not compiled, not run on hardware
//
#include <hip/hip_runtime.h>
#include <hip/hip_bf16.h>

typedef __attribute__((ext_vector_type(16))) __bf16 v16bf;
typedef __attribute__((ext_vector_type(8)))  float  v8f;

#define HWSZ 4096
#define BATCH 16

__device__ __forceinline__ unsigned short f2bf(float f) {
  union { float f; unsigned u; } c; c.f = f;
  unsigned u = c.u;
  unsigned r = u + 0x7FFFu + ((u >> 16) & 1u);   // round-to-nearest-even-ish
  return (unsigned short)(r >> 16);
}
__device__ __forceinline__ float bf2f(unsigned short h) {
  union { float f; unsigned u; } c; c.u = ((unsigned)h) << 16; return c.f;
}
__device__ __forceinline__ float sigm_(float x) { return 1.f / (1.f + __expf(-x)); }
__device__ __forceinline__ float silu_(float x) { return x * sigm_(x); }

// epilogue modes
#define EP_NONE       0
#define EP_BIAS       1
#define EP_BIAS_SILU  2
#define EP_BN         3
#define EP_BN_SILU    4
#define EP_BN_SIGMOID 5
#define EP_GATEMUL    6   // sigmoid(acc + pervec[b,co]) * aux[b,co,n]

#define BN_RSQ 0.9999950000374996f   // 1/sqrt(1+1e-5)

// ---------------------------------------------------------------------------
// Generic 1x1-conv GEMM:  Y[b, co, n] = sum_ci W[co, ci] * X[b, ci, n]
// bf16 WMMA 16x16x32, fp32 accumulate, double-buffered LDS, packed staging.
// X is bf16 (Xb) or fp32 (Xf). Output: fp32 (Yf) and/or bf16 (Yb) copies.
// Optional fp32 accumulator input (Yacc) for concat-input convs.
// Block = 256 threads = 8 waves; 64x64 tile; 2 wmma tiles per wave.
// ---------------------------------------------------------------------------
__global__ __launch_bounds__(256)
void gemm_wmma_k(const float* __restrict__ Wt, int ldw,
                 const unsigned short* __restrict__ Xb, const float* __restrict__ Xf,
                 const float* __restrict__ Yacc,
                 float* __restrict__ Yf, unsigned short* __restrict__ Yb,
                 int Cin, int Cout, int N,
                 const float* __restrict__ p0, const float* __restrict__ p1,
                 const float* __restrict__ pervec, const float* __restrict__ aux,
                 int mode) {
  const int n0 = blockIdx.x * 64;
  const int m0 = blockIdx.y * 64;
  const int b  = blockIdx.z;
  const long xbase = (long)b * Cin * N;
  const long ybase = (long)b * Cout * N;
  const int tid  = threadIdx.x;
  const int lane = tid & 31;
  const int wv   = tid >> 5;
  const int mW = (wv & 3) * 16;      // wave's 16-row slab of the 64x64 tile
  const int nW = (wv >> 2) * 32;     // wave's 32-col slab (two 16-col wmma tiles)

  __shared__ unsigned short lA[2][64 * 34];   // [row=cout][k], padded stride 34
  __shared__ unsigned short lB[2][64 * 34];   // [row=n   ][k], K-transposed

  // ---- staging: weights fp32 -> bf16 pairs, 2 packed dwords per item ----
  auto stageA = [&](int p, int k0) {
    #pragma unroll
    for (int i = 0; i < 2; ++i) {
      int id = tid + i * 256;                       // 0..511
      int r  = id >> 3;                             // 64 rows
      int kq = id & 7;                              // 8 float4 per row
      const float4 f4 = *(const float4*)&Wt[(long)(m0 + r) * ldw + k0 + kq * 4];
      unsigned d0 = (unsigned)f2bf(f4.x) | ((unsigned)f2bf(f4.y) << 16);
      unsigned d1 = (unsigned)f2bf(f4.z) | ((unsigned)f2bf(f4.w) << 16);
      unsigned* dst = (unsigned*)(lA[p] + r * 34 + kq * 4);
      dst[0] = d0; dst[1] = d1;
    }
  };
  // ---- staging: activations -> lB[n][k], packed dwords (pairs along k) ----
  auto stageB = [&](int p, int k0) {
    if (Xb) {
      #pragma unroll
      for (int i = 0; i < 2; ++i) {
        int id = tid + i * 256;                     // 0..511
        int n  = (id & 31) * 2;                     // n pair
        int kk = (id >> 5) * 2;                     // 16 k pairs
        unsigned u0 = *(const unsigned*)&Xb[xbase + (long)(k0 + kk) * N + n0 + n];
        unsigned u1 = *(const unsigned*)&Xb[xbase + (long)(k0 + kk + 1) * N + n0 + n];
        unsigned d0 = (u0 & 0xFFFFu) | (u1 << 16);          // row n
        unsigned d1 = (u0 >> 16) | (u1 & 0xFFFF0000u);      // row n+1
        *(unsigned*)(lB[p] + n * 34 + kk)       = d0;
        *(unsigned*)(lB[p] + (n + 1) * 34 + kk) = d1;
      }
    } else {
      #pragma unroll
      for (int i = 0; i < 4; ++i) {
        int id = tid + i * 256;                     // 0..1023
        int n  = id & 63;
        int kk = (id >> 6) * 2;
        float a = Xf[xbase + (long)(k0 + kk) * N + n0 + n];
        float c = Xf[xbase + (long)(k0 + kk + 1) * N + n0 + n];
        *(unsigned*)(lB[p] + n * 34 + kk) =
            (unsigned)f2bf(a) | ((unsigned)f2bf(c) << 16);
      }
    }
  };

  v8f acc0 = {}; v8f acc1 = {};
  stageA(0, 0); stageB(0, 0);
  int p = 0;
  for (int k0 = 0; k0 < Cin; k0 += 32) {
    __syncthreads();
    if (k0 + 32 < Cin) { stageA(p ^ 1, k0 + 32); stageB(p ^ 1, k0 + 32); }
    // A fragment (16x32 bf16): M=lane&15; kbase=(lane>>4)*8; pairs per VGPR
    union Frag { v16bf v; unsigned u[8]; };
    Frag fa, fb0, fb1;
    const int am  = mW + (lane & 15);
    const int akb = (lane >> 4) * 8;
    #pragma unroll
    for (int j = 0; j < 8; ++j) {
      int kk = akb + (j >> 2) * 16 + (j & 3) * 2;
      fa.u[j] = *(const unsigned*)(lA[p] + am * 34 + kk);
    }
    // B fragments (32x16 bf16): n=lane&15; kbase=(lane>>4)*8; same k order as A
    const int bn  = lane & 15;
    const int bkb = (lane >> 4) * 8;
    #pragma unroll
    for (int j = 0; j < 8; ++j) {
      int kk = bkb + (j >> 2) * 16 + (j & 3) * 2;
      fb0.u[j] = *(const unsigned*)(lB[p] + (nW +      bn) * 34 + kk);
      fb1.u[j] = *(const unsigned*)(lB[p] + (nW + 16 + bn) * 34 + kk);
    }
    acc0 = __builtin_amdgcn_wmma_f32_16x16x32_bf16(false, fa.v, false, fb0.v,
                                                   (short)0, acc0, false, false);
    acc1 = __builtin_amdgcn_wmma_f32_16x16x32_bf16(false, fa.v, false, fb1.v,
                                                   (short)0, acc1, false, false);
    p ^= 1;
  }

  // epilogue: C/D layout -> n = lane&15, m = (lane>>4)*8 + j
  const int cn    = lane & 15;
  const int mhalf = (lane >> 4) * 8;
  #pragma unroll
  for (int j = 0; j < 8; ++j) {
    int m  = mW + mhalf + j;
    int co = m0 + m;
    long o0 = ybase + (long)co * N + n0 + nW + cn;
    long o1 = o0 + 16;
    float v0 = acc0[j], v1 = acc1[j];
    if (Yacc) { v0 += Yacc[o0]; v1 += Yacc[o1]; }
    switch (mode) {
      case EP_BIAS:       v0 += p0[co]; v1 += p0[co]; break;
      case EP_BIAS_SILU:  v0 = silu_(v0 + p0[co]); v1 = silu_(v1 + p0[co]); break;
      case EP_BN: {
        float s = p0[co] * BN_RSQ, t = p1[co];
        v0 = v0 * s + t; v1 = v1 * s + t; } break;
      case EP_BN_SILU: {
        float s = p0[co] * BN_RSQ, t = p1[co];
        v0 = silu_(v0 * s + t); v1 = silu_(v1 * s + t); } break;
      case EP_BN_SIGMOID: {
        float s = p0[co] * BN_RSQ, t = p1[co];
        v0 = sigm_(v0 * s + t); v1 = sigm_(v1 * s + t); } break;
      case EP_GATEMUL: {
        float pv = pervec[(long)b * Cout + co];
        v0 = sigm_(v0 + pv) * aux[o0];
        v1 = sigm_(v1 + pv) * aux[o1]; } break;
      default: break;
    }
    if (Yf) { Yf[o0] = v0; Yf[o1] = v1; }
    if (Yb) { Yb[o0] = f2bf(v0); Yb[o1] = f2bf(v1); }
  }
}

// ---------------------------------------------------------------------------
// mean over H,W: g[b,c] = mean(x[b,c,:,:])
// ---------------------------------------------------------------------------
__global__ __launch_bounds__(256)
void mean_hw_k(const float* __restrict__ X, float* __restrict__ out, int C) {
  int c = blockIdx.x, b = blockIdx.y;
  const float* p = X + ((long)b * C + c) * HWSZ;
  float s = 0.f;
  for (int i = threadIdx.x; i < HWSZ; i += 256) s += p[i];
  __shared__ float red[256];
  red[threadIdx.x] = s; __syncthreads();
  for (int st = 128; st > 0; st >>= 1) {
    if (threadIdx.x < st) red[threadIdx.x] += red[threadIdx.x + st];
    __syncthreads();
  }
  if (threadIdx.x == 0) out[(long)b * C + c] = red[0] * (1.f / HWSZ);
}

// ---------------------------------------------------------------------------
// global-squeeze MLP + per-batch fg contribution:
// gv = W2 @ silu(W1 @ g + b1) + b2 ;  gfc[b,co] = fg_b[co] + fg_w[co,0:256]·gv
// ---------------------------------------------------------------------------
__global__ __launch_bounds__(256)
void squeeze_mlp_k(const float* __restrict__ gmean,
                   const float* __restrict__ w1, const float* __restrict__ b1,
                   const float* __restrict__ w2, const float* __restrict__ b2,
                   const float* __restrict__ fgw, const float* __restrict__ fgb,
                   float* __restrict__ gfc) {
  int b = blockIdx.x, t = threadIdx.x;
  __shared__ float gin[256], hid[64], gv[256];
  gin[t] = gmean[(long)b * 256 + t];
  __syncthreads();
  if (t < 64) {
    float s = b1[t];
    for (int ci = 0; ci < 256; ++ci) s += w1[t * 256 + ci] * gin[ci];
    hid[t] = silu_(s);
  }
  __syncthreads();
  {
    float s = b2[t];
    for (int h = 0; h < 64; ++h) s += w2[t * 64 + h] * hid[h];
    gv[t] = s;
  }
  __syncthreads();
  float s = fgb[t];
  for (int ci = 0; ci < 256; ++ci) s += fgw[(long)t * 768 + ci] * gv[ci];
  gfc[(long)b * 256 + t] = s;
}

// ---------------------------------------------------------------------------
// grouped 3x3 conv (groups=64, 4 in-ch/group) + bias + silu -> bf16 (ls branch)
// ---------------------------------------------------------------------------
__global__ __launch_bounds__(256)
void lsconv_k(const float* __restrict__ x, const float* __restrict__ w,
              const float* __restrict__ bias, unsigned short* __restrict__ out) {
  int pix = blockIdx.x * 256 + threadIdx.x;
  int oc = blockIdx.y, b = blockIdx.z;
  int h = pix >> 6, wi = pix & 63;
  float s = bias[oc];
  for (int j = 0; j < 4; ++j) {
    const float* xp = x + ((long)b * 256 + oc * 4 + j) * HWSZ;
    const float* wp = w + (oc * 4 + j) * 9;
    #pragma unroll
    for (int dy = -1; dy <= 1; ++dy) {
      int hh = h + dy; if (hh < 0 || hh > 63) continue;
      #pragma unroll
      for (int dx = -1; dx <= 1; ++dx) {
        int ww = wi + dx; if (ww < 0 || ww > 63) continue;
        s += wp[(dy + 1) * 3 + dx + 1] * xp[hh * 64 + ww];
      }
    }
  }
  out[((long)b * 64 + oc) * HWSZ + pix] = f2bf(silu_(s));
}

// ---------------------------------------------------------------------------
// dense 3x3 conv 64->64 + bias + silu -> bf16 (be branch middle)
// ---------------------------------------------------------------------------
__global__ __launch_bounds__(256)
void conv3x3_64_k(const float* __restrict__ x, const float* __restrict__ w,
                  const float* __restrict__ bias, unsigned short* __restrict__ out) {
  int pix = blockIdx.x * 256 + threadIdx.x;
  int oc = blockIdx.y, b = blockIdx.z;
  int h = pix >> 6, wi = pix & 63;
  float s = bias[oc];
  for (int ci = 0; ci < 64; ++ci) {
    const float* xp = x + ((long)b * 64 + ci) * HWSZ;
    const float* wp = w + (oc * 64 + ci) * 9;
    #pragma unroll
    for (int dy = -1; dy <= 1; ++dy) {
      int hh = h + dy; if (hh < 0 || hh > 63) continue;
      #pragma unroll
      for (int dx = -1; dx <= 1; ++dx) {
        int ww = wi + dx; if (ww < 0 || ww > 63) continue;
        s += wp[(dy + 1) * 3 + dx + 1] * xp[hh * 64 + ww];
      }
    }
  }
  out[((long)b * 64 + oc) * HWSZ + pix] = f2bf(silu_(s));
}

// ---------------------------------------------------------------------------
// depthwise 3x3 over concat(q,k,v) + BN + silu -> bf16 (detail enhance stage 1)
// ---------------------------------------------------------------------------
__global__ __launch_bounds__(256)
void de_dw_k(const float* __restrict__ q, const float* __restrict__ k,
             const float* __restrict__ v, const float* __restrict__ w,
             const float* __restrict__ g, const float* __restrict__ bt,
             unsigned short* __restrict__ out) {
  int pix = blockIdx.x * 256 + threadIdx.x;
  int c = blockIdx.y, b = blockIdx.z;
  int h = pix >> 6, wi = pix & 63;
  const float* src = (c < 64)  ? q + ((long)b * 64  + c)        * HWSZ
                   : (c < 128) ? k + ((long)b * 64  + (c - 64)) * HWSZ
                               : v + ((long)b * 128 + (c - 128)) * HWSZ;
  const float* wp = w + c * 9;
  float s = 0.f;
  #pragma unroll
  for (int dy = -1; dy <= 1; ++dy) {
    int hh = h + dy; if (hh < 0 || hh > 63) continue;
    #pragma unroll
    for (int dx = -1; dx <= 1; ++dx) {
      int ww = wi + dx; if (ww < 0 || ww > 63) continue;
      s += wp[(dy + 1) * 3 + dx + 1] * src[hh * 64 + ww];
    }
  }
  s = s * (g[c] * BN_RSQ) + bt[c];
  out[((long)b * 256 + c) * HWSZ + pix] = f2bf(silu_(s));
}

// ---------------------------------------------------------------------------
// axial mean: rowmode=1 -> mean over W (thread=h); else mean over H (thread=w)
// ---------------------------------------------------------------------------
__global__ __launch_bounds__(64)
void axial_mean_k(const float* __restrict__ X, float* __restrict__ out,
                  int C, int rowmode) {
  int c = blockIdx.x, b = blockIdx.y, t = threadIdx.x;
  const float* p = X + ((long)b * C + c) * HWSZ;
  float s = 0.f;
  if (rowmode) { for (int w = 0; w < 64; ++w) s += p[t * 64 + w]; }
  else         { for (int h = 0; h < 64; ++h) s += p[h * 64 + t]; }
  out[((long)b * C + c) * 64 + t] = s * (1.f / 64.f);
}

// ---------------------------------------------------------------------------
// axial softmax attention per (head, batch); L=64; pe interp is identity.
// writes bf16 silu(attn @ v) (feeds the following WMMA projection)
// ---------------------------------------------------------------------------
__global__ __launch_bounds__(64)
void axial_attn_k(const float* __restrict__ qm, const float* __restrict__ km,
                  const float* __restrict__ vm, const float* __restrict__ peq,
                  const float* __restrict__ pek, unsigned short* __restrict__ out) {
  int n = blockIdx.x, b = blockIdx.y, i = threadIdx.x;
  __shared__ float qs[16][64], ks[16][64], vs[32][64], ss[64][65];
  for (int d = 0; d < 16; ++d) {
    qs[d][i] = qm[((long)b * 64 + n * 16 + d) * 64 + i] + peq[(n * 16 + d) * 64 + i];
    ks[d][i] = km[((long)b * 64 + n * 16 + d) * 64 + i] + pek[(n * 16 + d) * 64 + i];
  }
  for (int d = 0; d < 32; ++d)
    vs[d][i] = vm[((long)b * 128 + n * 32 + d) * 64 + i];
  __syncthreads();
  float qi[16];
  #pragma unroll
  for (int d = 0; d < 16; ++d) qi[d] = qs[d][i];
  float mx = -1e30f;
  for (int j = 0; j < 64; ++j) {
    float s = 0.f;
    #pragma unroll
    for (int d = 0; d < 16; ++d) s += qi[d] * ks[d][j];
    s *= 0.25f;                       // 16^-0.5, temperature 1
    ss[i][j] = s; mx = fmaxf(mx, s);
  }
  float den = 0.f;
  for (int j = 0; j < 64; ++j) { float e = __expf(ss[i][j] - mx); ss[i][j] = e; den += e; }
  float inv = 1.f / den;
  for (int d = 0; d < 32; ++d) {
    float o = 0.f;
    for (int j = 0; j < 64; ++j) o += ss[i][j] * vs[d][j];
    o *= inv;
    out[((long)b * 128 + n * 32 + d) * 64 + i] = f2bf(silu_(o));
  }
}

// fw[b,c] = sigmoid(mean_h rowp[b,c,h])
__global__ __launch_bounds__(128)
void fuse_weight_k(const float* __restrict__ rowp, float* __restrict__ fw) {
  int b = blockIdx.x, c = threadIdx.x;
  float s = 0.f;
  for (int h = 0; h < 64; ++h) s += rowp[((long)b * 128 + c) * 64 + h];
  fw[(long)b * 128 + c] = sigm_(s * (1.f / 64.f));
}

// enhanced = v + fw*row + (1-fw)*col -> bf16 (feeds op WMMA projection)
__global__ __launch_bounds__(256)
void fuse_axial_k(const float* __restrict__ v, const float* __restrict__ rowp,
                  const float* __restrict__ colp, const float* __restrict__ fw,
                  unsigned short* __restrict__ out) {
  int pix = blockIdx.x * 256 + threadIdx.x;
  int c = blockIdx.y, b = blockIdx.z;
  int h = pix >> 6, w = pix & 63;
  float f = fw[(long)b * 128 + c];
  long base = ((long)b * 128 + c);
  float val = v[base * HWSZ + pix]
            + f * rowp[base * 64 + h]
            + (1.f - f) * colp[base * 64 + w];
  out[base * HWSZ + pix] = f2bf(val);
}

// ---------------------------------------------------------------------------
// final: mix = out*gate + de*(1-gate) + 0.1*x, then channel LayerNorm (C=256).
// 64 threads = 64 pixels per block; pixel's 256-ch vector stashed in LDS.
// ---------------------------------------------------------------------------
__global__ __launch_bounds__(64)
void final_ln_k(const float* __restrict__ outb, const float* __restrict__ de,
                const float* __restrict__ gate, const float* __restrict__ x,
                const float* __restrict__ lng, const float* __restrict__ lnb,
                float* __restrict__ y) {
  int b = blockIdx.y, t = threadIdx.x;
  int pix = blockIdx.x * 64 + t;
  __shared__ float stash[256 * 64];
  float s = 0.f, s2 = 0.f;
  for (int c = 0; c < 256; ++c) {
    long idx = ((long)b * 256 + c) * HWSZ + pix;
    float o = outb[idx], g = gate[idx], d = de[idx], xv = x[idx];
    float val = o * g + d * (1.f - g) + 0.1f * xv;
    stash[c * 64 + t] = val;
    s += val; s2 += val * val;
  }
  float mu  = s * (1.f / 256.f);
  float var = s2 * (1.f / 256.f) - mu * mu;
  float inv = rsqrtf(var + 1e-5f);
  for (int c = 0; c < 256; ++c) {
    long idx = ((long)b * 256 + c) * HWSZ + pix;
    y[idx] = (stash[c * 64 + t] - mu) * inv * lng[c] + lnb[c];
  }
}

// ---------------------------------------------------------------------------
static void gemm(hipStream_t s, const float* Wt, int ldw,
                 const unsigned short* Xb, const float* Xf,
                 const float* Yacc, float* Yf, unsigned short* Yb,
                 int Cin, int Cout, int N,
                 const float* p0, const float* p1,
                 const float* pervec, const float* aux, int mode) {
  dim3 g(N / 64, Cout / 64, BATCH);
  gemm_wmma_k<<<g, 256, 0, s>>>(Wt, ldw, Xb, Xf, Yacc, Yf, Yb, Cin, Cout, N,
                                p0, p1, pervec, aux, mode);
}

extern "C" void kernel_launch(void* const* d_in, const int* in_sizes, int n_in,
                              void* d_out, int out_size, void* d_ws, size_t ws_size,
                              hipStream_t stream) {
  (void)in_sizes; (void)n_in; (void)out_size; (void)ws_size;
  #define P(i) ((const float*)d_in[i])
  const float* x = P(0);
  const float *gs_w1 = P(1), *gs_b1 = P(2), *gs_w2 = P(3), *gs_b2 = P(4);
  const float *ls_dw = P(5), *ls_b1 = P(6), *ls_w2 = P(7), *ls_b2 = P(8);
  const float *be_w1 = P(9), *be_b1 = P(10), *be_w2 = P(11), *be_b2 = P(12);
  const float *be_w3 = P(13), *be_b3 = P(14);
  const float *fg_w = P(15), *fg_b = P(16);
  const float *wq = P(17), *qg = P(18), *qb = P(19);
  const float *wk = P(20), *kg = P(21), *kb = P(22);
  const float *wv = P(23), *vg = P(24), *vb = P(25);
  const float *de_dw = P(26), *de_g1 = P(27), *de_b1 = P(28);
  const float *de_w2 = P(29), *de_g2 = P(30), *de_b2 = P(31);
  const float *pe_rq = P(32), *pe_rk = P(33), *pe_cq = P(34), *pe_ck = P(35);
  const float *rp_w = P(36), *rp_g = P(37), *rp_b = P(38);
  const float *cp_w = P(39), *cp_g = P(40), *cp_b = P(41);
  const float *op_w = P(42), *op_g = P(43), *op_b = P(44);
  const float *dg_w1 = P(45), *dg_g1 = P(46), *dg_b1 = P(47);
  const float *dg_w2 = P(48), *dg_g2 = P(49), *dg_b2 = P(50);
  const float *ln_g = P(51), *ln_b = P(52);
  #undef P

  // ---- workspace carve ----
  const size_t N64  = (size_t)BATCH * 64  * HWSZ;
  const size_t N128 = (size_t)BATCH * 128 * HWSZ;
  const size_t N256 = (size_t)BATCH * 256 * HWSZ;
  char* base = (char*)d_ws;
  auto AF = [&](size_t n) { float* p = (float*)base; base += n * 4; return p; };
  auto AB = [&](size_t n) { unsigned short* p = (unsigned short*)base; base += n * 2; return p; };

  unsigned short* SB0b = AB(N64);    // lhid_b -> b2_b
  unsigned short* BFAb = AB(N256);   // l_b -> de1_b
  unsigned short* BFBb = AB(N256);   // bdy_b -> gtmp_b
  unsigned short* BFCb = AB(N256);   // xs_b -> enh_b
  float* ACCf = AF(N256);            // fg partial -> dg partial -> gate
  float* B1f  = AF(N64);             // be hidden 1 (fp32, VALU consumer)
  float* Qf   = AF(N64);
  float* Kf   = AF(N64);
  float* Vf   = AF(N128);
  float* DEf  = AF(N256);  unsigned short* DEb  = AB(N256);
  float* OUTf = AF(N256);  unsigned short* OUTb = AB(N256);
  float* gmean = AF(4096);
  float* gfc   = AF(4096);
  float* qr = AF(65536);  float* kr = AF(65536);  float* vr = AF(131072);
  float* qc = AF(65536);  float* kc = AF(65536);  float* vc = AF(131072);
  unsigned short* rowa = AB(131072); unsigned short* cola = AB(131072);
  float* rowp = AF(131072); float* colp = AF(131072);
  float* fwv  = AF(2048);

  // --- squeeze module ---
  mean_hw_k<<<dim3(256, BATCH), 256, 0, stream>>>(x, gmean, 256);
  squeeze_mlp_k<<<BATCH, 256, 0, stream>>>(gmean, gs_w1, gs_b1, gs_w2, gs_b2,
                                           fg_w, fg_b, gfc);
  lsconv_k<<<dim3(16, 64, BATCH), 256, 0, stream>>>(x, ls_dw, ls_b1, SB0b);
  gemm(stream, ls_w2, 64, SB0b, nullptr, nullptr, nullptr, BFAb,
       64, 256, HWSZ, ls_b2, nullptr, nullptr, nullptr, EP_BIAS);          // l (bf16)
  gemm(stream, be_w1, 256, nullptr, x, nullptr, B1f, nullptr,
       256, 64, HWSZ, be_b1, nullptr, nullptr, nullptr, EP_BIAS_SILU);
  conv3x3_64_k<<<dim3(16, 64, BATCH), 256, 0, stream>>>(B1f, be_w2, be_b2, SB0b);
  gemm(stream, be_w3, 64, SB0b, nullptr, nullptr, nullptr, BFBb,
       64, 256, HWSZ, be_b3, nullptr, nullptr, nullptr, EP_BIAS);          // bdy (bf16)
  // gate0 = sigmoid(fg_w·[g;l;bdy]+b);  xs = x * gate0 (fully fused)
  gemm(stream, fg_w + 256, 768, BFAb, nullptr, nullptr, ACCf, nullptr,
       256, 256, HWSZ, nullptr, nullptr, nullptr, nullptr, EP_NONE);
  gemm(stream, fg_w + 512, 768, BFBb, nullptr, ACCf, nullptr, BFCb,
       256, 256, HWSZ, nullptr, nullptr, gfc, x, EP_GATEMUL);              // xs (bf16)

  // --- q, k, v projections (BN epilogue, fp32 for VALU consumers) ---
  gemm(stream, wq, 256, BFCb, nullptr, nullptr, Qf, nullptr,
       256, 64, HWSZ, qg, qb, nullptr, nullptr, EP_BN);
  gemm(stream, wk, 256, BFCb, nullptr, nullptr, Kf, nullptr,
       256, 64, HWSZ, kg, kb, nullptr, nullptr, EP_BN);
  gemm(stream, wv, 256, BFCb, nullptr, nullptr, Vf, nullptr,
       256, 128, HWSZ, vg, vb, nullptr, nullptr, EP_BN);

  // --- detail enhancement ---
  de_dw_k<<<dim3(16, 256, BATCH), 256, 0, stream>>>(Qf, Kf, Vf, de_dw, de_g1,
                                                    de_b1, BFAb);          // de1 (bf16)
  gemm(stream, de_w2, 256, BFAb, nullptr, nullptr, DEf, DEb,
       256, 256, HWSZ, de_g2, de_b2, nullptr, nullptr, EP_BN);             // de (dual)

  // --- axial attention ---
  axial_mean_k<<<dim3(64, BATCH), 64, 0, stream>>>(Qf, qr, 64, 1);
  axial_mean_k<<<dim3(64, BATCH), 64, 0, stream>>>(Kf, kr, 64, 1);
  axial_mean_k<<<dim3(128, BATCH), 64, 0, stream>>>(Vf, vr, 128, 1);
  axial_mean_k<<<dim3(64, BATCH), 64, 0, stream>>>(Qf, qc, 64, 0);
  axial_mean_k<<<dim3(64, BATCH), 64, 0, stream>>>(Kf, kc, 64, 0);
  axial_mean_k<<<dim3(128, BATCH), 64, 0, stream>>>(Vf, vc, 128, 0);
  axial_attn_k<<<dim3(4, BATCH), 64, 0, stream>>>(qr, kr, vr, pe_rq, pe_rk, rowa);
  axial_attn_k<<<dim3(4, BATCH), 64, 0, stream>>>(qc, kc, vc, pe_cq, pe_ck, cola);
  gemm(stream, rp_w, 128, rowa, nullptr, nullptr, rowp, nullptr,
       128, 128, 64, rp_g, rp_b, nullptr, nullptr, EP_BN);
  gemm(stream, cp_w, 128, cola, nullptr, nullptr, colp, nullptr,
       128, 128, 64, cp_g, cp_b, nullptr, nullptr, EP_BN);
  fuse_weight_k<<<BATCH, 128, 0, stream>>>(rowp, fwv);
  fuse_axial_k<<<dim3(16, 128, BATCH), 256, 0, stream>>>(Vf, rowp, colp, fwv,
                                                         BFCb);            // enh (bf16)

  // --- output projection + detection gate ---
  gemm(stream, op_w, 128, BFCb, nullptr, nullptr, OUTf, OUTb,
       128, 256, HWSZ, op_g, op_b, nullptr, nullptr, EP_BN);               // out (dual)
  gemm(stream, dg_w1, 512, OUTb, nullptr, nullptr, ACCf, nullptr,
       256, 256, HWSZ, nullptr, nullptr, nullptr, nullptr, EP_NONE);
  gemm(stream, dg_w1 + 256, 512, DEb, nullptr, ACCf, nullptr, BFBb,
       256, 256, HWSZ, dg_g1, dg_b1, nullptr, nullptr, EP_BN_SILU);        // dg hidden
  gemm(stream, dg_w2, 256, BFBb, nullptr, nullptr, ACCf, nullptr,
       256, 256, HWSZ, dg_g2, dg_b2, nullptr, nullptr, EP_BN_SIGMOID);     // gate

  // --- final mix + LayerNorm ---
  final_ln_k<<<dim3(64, BATCH), 64, 0, stream>>>(OUTf, DEf, ACCf, x, ln_g, ln_b,
                                                 (float*)d_out);
}
